// MultiHeadCrossAttention_51161650430221
// MI455X (gfx1250) — matrix operand order repaired and checked by reading, not compiled, not run on hardware
//
#include <hip/hip_runtime.h>
#include <cstdint>

typedef unsigned short u16;
typedef unsigned int   u32;
typedef __attribute__((ext_vector_type(16))) __bf16 bf16x16;
typedef __attribute__((ext_vector_type(8)))  float  v8f;

// ---------- helpers ----------

__device__ __forceinline__ u16 f2bf(float x) {
    u32 u = __float_as_uint(x);
    u32 r = u + 0x7FFFu + ((u >> 16) & 1u);   // round-to-nearest-even
    return (u16)(r >> 16);
}

union FragU {
    bf16x16 v;
    uint4   q[2];
    u16     u[16];
};

// A-matrix 16x32 bf16 fragment from row-major-in-K LDS tile.
// Lane L: row = roff + (L&15); lanes 0-15 hold K {0..7,16..23}, lanes 16-31 hold K {8..15,24..31}.
__device__ __forceinline__ bf16x16 load_a_frag(const u16* lds, int stride, int roff, int koff, int lane) {
    int r  = roff + (lane & 15);
    int kb = koff + ((lane >> 4) << 3);            // 0 or 8
    FragU f;
    f.q[0] = *(const uint4*)(lds + r * stride + kb);
    f.q[1] = *(const uint4*)(lds + r * stride + kb + 16);
    return f.v;
}

// B-matrix 32x16 bf16 fragment from LDS tile stored [N][K] row-major.
// Lane L: col = coff + (L&15); lanes 0-15 hold K {0..7,16..23}, lanes 16-31 hold K {8..15,24..31}.
__device__ __forceinline__ bf16x16 load_b_frag(const u16* lds, int stride, int coff, int koff, int lane) {
    int c  = coff + (lane & 15);
    int kb = koff + ((lane >> 4) << 3);            // 0 or 8
    FragU f;
    f.q[0] = *(const uint4*)(lds + c * stride + kb);
    f.q[1] = *(const uint4*)(lds + c * stride + kb + 16);
    return f.v;
}

__device__ __forceinline__ v8f wmma_bf16(bf16x16 a, bf16x16 b, v8f c) {
    return __builtin_amdgcn_wmma_f32_16x16x32_bf16(false, a, false, b, (short)0, c, false, false);
}

__device__ __forceinline__ void store_c(float* p, float v) { *p = v; }
__device__ __forceinline__ void store_c(u16*   p, float v) { *p = f2bf(v); }

// ---------- CDNA5 async global->LDS copy (ASYNCcnt-tracked, no VGPR staging) ----------

__device__ __forceinline__ u32 lds_byte_off(const void* p) {
    // addrspace(3) / generic LDS addresses carry the byte offset in the low 32 bits
    return (u32)(uintptr_t)p;
}

__device__ __forceinline__ void async_ld_b128(u32 lds, const u16* g) {
    unsigned long long ga = (unsigned long long)(uintptr_t)g;
    asm volatile("global_load_async_to_lds_b128 %0, %1, off"
                 :: "v"(lds), "v"(ga) : "memory");
}

#if __has_builtin(__builtin_amdgcn_s_wait_asynccnt)
#define WAIT_ASYNC(n) __builtin_amdgcn_s_wait_asynccnt((unsigned short)(n))
#else
#define WAIT_ASYNC(n) asm volatile("s_wait_asynccnt %0" :: "i"(n) : "memory")
#endif

// ---------- fp32 -> bf16 conversion ----------

__global__ __launch_bounds__(256) void cvt_bf16(const float* __restrict__ in,
                                                u16* __restrict__ out, int n4) {
    int i = blockIdx.x * blockDim.x + threadIdx.x;
    if (i >= n4) return;
    float4 v = ((const float4*)in)[i];
    uint2 o;
    o.x = (u32)f2bf(v.x) | ((u32)f2bf(v.y) << 16);
    o.y = (u32)f2bf(v.z) | ((u32)f2bf(v.w) << 16);
    ((uint2*)out)[i] = o;
}

// ---------- bf16 GEMM: C[M,N] = A[M,K] @ Bw[N,K]^T (fp32 accumulate) ----------
// Block tile 128x128x32, 256 threads (8 waves), wave tile 32x64 (2x4 WMMA tiles).
// Double-buffered LDS fed by async-to-LDS DMA; prefetch overlaps the WMMA block.

template <typename OutT>
__global__ __launch_bounds__(256) void gemm_bf16_nt(const u16* __restrict__ A,
                                                    const u16* __restrict__ Bw,
                                                    OutT* __restrict__ C,
                                                    int M, int N, int K) {
    __shared__ __align__(16) u16 As[2][128 * 32];
    __shared__ __align__(16) u16 Bs[2][128 * 32];

    const int t    = threadIdx.x;
    const int lane = t & 31;
    const int wave = t >> 5;
    const int m0   = blockIdx.y * 128;
    const int n0   = blockIdx.x * 128;
    const int mo   = (wave >> 1) * 32;   // 0,32,64,96
    const int no   = (wave & 1) * 64;    // 0,64

    const int rowL = t >> 1;             // 0..127
    const int segL = (t & 1) * 16;       // 0 or 16 halves

    const u16* gA = A  + (size_t)(m0 + rowL) * K + segL;
    const u16* gB = Bw + (size_t)(n0 + rowL) * K + segL;
    const u32 lA[2] = { lds_byte_off(&As[0][rowL * 32 + segL]),
                        lds_byte_off(&As[1][rowL * 32 + segL]) };
    const u32 lB[2] = { lds_byte_off(&Bs[0][rowL * 32 + segL]),
                        lds_byte_off(&Bs[1][rowL * 32 + segL]) };

    // prefetch k-tile 0 into buffer 0 (4 x b128 per thread)
    async_ld_b128(lA[0],      gA);
    async_ld_b128(lA[0] + 16, gA + 8);
    async_ld_b128(lB[0],      gB);
    async_ld_b128(lB[0] + 16, gB + 8);

    v8f acc[2][4];
#pragma unroll
    for (int i = 0; i < 2; ++i)
#pragma unroll
        for (int j = 0; j < 4; ++j)
            acc[i][j] = (v8f){0.f, 0.f, 0.f, 0.f, 0.f, 0.f, 0.f, 0.f};

    const int nIter = K / 32;
    for (int it = 0; it < nIter; ++it) {
        const int cur = it & 1;
        const int nxt = cur ^ 1;

        __syncthreads();   // all waves done reading buf[nxt] (iteration it-1) -> safe to overwrite
        if (it + 1 < nIter) {
            const u16* gA2 = gA + (size_t)(it + 1) * 32;
            const u16* gB2 = gB + (size_t)(it + 1) * 32;
            async_ld_b128(lA[nxt],      gA2);
            async_ld_b128(lA[nxt] + 16, gA2 + 8);
            async_ld_b128(lB[nxt],      gB2);
            async_ld_b128(lB[nxt] + 16, gB2 + 8);
            WAIT_ASYNC(4);   // newest 4 (prefetch) may remain; current tile's 4 are done
        } else {
            WAIT_ASYNC(0);
        }
        __syncthreads();   // current buffer visible to all waves

        const u16* Ac = As[cur];
        const u16* Bc = Bs[cur];
        bf16x16 af0 = load_a_frag(Ac, 32, mo,      0, lane);
        bf16x16 af1 = load_a_frag(Ac, 32, mo + 16, 0, lane);
#pragma unroll
        for (int j = 0; j < 4; ++j) {
            bf16x16 bf = load_b_frag(Bc, 32, no + j * 16, 0, lane);
            acc[0][j] = wmma_bf16(af0, bf, acc[0][j]);
            acc[1][j] = wmma_bf16(af1, bf, acc[1][j]);
        }
    }

    const int mrow = (lane >> 4) << 3;   // 0 or 8
    const int ncol = lane & 15;
#pragma unroll
    for (int i2 = 0; i2 < 2; ++i2)
#pragma unroll
        for (int j = 0; j < 4; ++j)
#pragma unroll
            for (int r = 0; r < 8; ++r) {
                int m = m0 + mo + i2 * 16 + r + mrow;
                int n = n0 + no + j * 16 + ncol;
                store_c(&C[(size_t)m * N + n], acc[i2][j][r]);
            }
}

// ---------- streaming attention (naive softmax, single pass) ----------
// Grid: (S/64, B*H). Block 256 = 8 waves. Per block: 64 queries x full KV sweep.

__global__ __launch_bounds__(256) void attn64(const u16* __restrict__ Qb,
                                              const u16* __restrict__ Kb,
                                              const u16* __restrict__ Vb,
                                              const float* __restrict__ mask,
                                              u16* __restrict__ Ob) {
    constexpr int S  = 2048;
    constexpr int H  = 16;
    constexpr int Dm = 1024;

    __shared__ __align__(16) u16 Qs[64 * 64];
    __shared__ __align__(16) u16 Ks[64 * 64];
    __shared__ __align__(16) u16 Vt[64 * 64];   // transposed: [dim][kv]
    __shared__ __align__(16) u16 Ps[64 * 64];
    __shared__ float rowsum[64];

    const int t    = threadIdx.x;
    const int lane = t & 31;
    const int wave = t >> 5;
    const int b    = blockIdx.y / H;
    const int h    = blockIdx.y % H;
    const int q0   = blockIdx.x * 64;
    const int mo   = (wave >> 1) * 16;   // 0,16,32,48
    const int no   = (wave & 1) * 32;    // 0,32

    if (t < 64) rowsum[t] = 0.0f;

    v8f oacc[2];
    oacc[0] = (v8f){0.f, 0.f, 0.f, 0.f, 0.f, 0.f, 0.f, 0.f};
    oacc[1] = (v8f){0.f, 0.f, 0.f, 0.f, 0.f, 0.f, 0.f, 0.f};

    const int row = t >> 2;              // 0..63
    const int seg = (t & 3) * 16;        // 0,16,32,48 halves

    {   // load Q tile once
        const uint4* g = (const uint4*)(Qb + (size_t)(b * S + q0 + row) * Dm + h * 64 + seg);
        uint4 v0 = g[0], v1 = g[1];
        *(uint4*)&Qs[row * 64 + seg]     = v0;
        *(uint4*)&Qs[row * 64 + seg + 8] = v1;
    }

    const int mrow = (lane >> 4) << 3;
    const int ncol = lane & 15;
    const float scale = 0.125f;          // 1/sqrt(64)

    // hoisted row pointers: only adds inside the KV loop
    const u16* kPtr = Kb + (size_t)(b * S + row) * Dm + h * 64 + seg;
    const u16* vPtr = Vb + (size_t)(b * S + row) * Dm + h * 64 + seg;
    const size_t kvStep = (size_t)64 * Dm;
    const float* mp[8];
#pragma unroll
    for (int i = 0; i < 8; ++i)
        mp[i] = mask + (size_t)(q0 + mo + i + mrow) * S + no + ncol;

    for (int kv0 = 0; kv0 < S; kv0 += 64) {
        const uint4* gk = (const uint4*)kPtr;
        const uint4* gv = (const uint4*)vPtr;
        uint4 k0 = gk[0], k1 = gk[1];
        uint4 v0 = gv[0], v1 = gv[1];
        kPtr += kvStep;
        vPtr += kvStep;

        __syncthreads();  // previous iter readers done (also fences the Q-tile write on iter 0)
        *(uint4*)&Ks[row * 64 + seg]     = k0;
        *(uint4*)&Ks[row * 64 + seg + 8] = k1;
        {   // V transposed into LDS so its B-fragments are contiguous
            union { uint4 q[2]; u16 u[16]; } tv;
            tv.q[0] = v0; tv.q[1] = v1;
#pragma unroll
            for (int j = 0; j < 16; ++j) Vt[(seg + j) * 64 + row] = tv.u[j];
        }
        __syncthreads();

        // S = Q K^T  (K dim = 64 -> two k-chunks of 32)
        bf16x16 qa0 = load_a_frag(Qs, 64, mo, 0,  lane);
        bf16x16 qa1 = load_a_frag(Qs, 64, mo, 32, lane);
        v8f sacc[2];
        sacc[0] = (v8f){0.f, 0.f, 0.f, 0.f, 0.f, 0.f, 0.f, 0.f};
        sacc[1] = (v8f){0.f, 0.f, 0.f, 0.f, 0.f, 0.f, 0.f, 0.f};
#pragma unroll
        for (int tj = 0; tj < 2; ++tj) {
            bf16x16 kb0 = load_b_frag(Ks, 64, no + tj * 16, 0,  lane);
            sacc[tj] = wmma_bf16(qa0, kb0, sacc[tj]);
            bf16x16 kb1 = load_b_frag(Ks, 64, no + tj * 16, 32, lane);
            sacc[tj] = wmma_bf16(qa1, kb1, sacc[tj]);
        }

        // p = exp(s/8 + mask); accumulate fp32 row sums; stage bf16 P in LDS
#pragma unroll
        for (int tj = 0; tj < 2; ++tj) {
#pragma unroll
            for (int i = 0; i < 8; ++i) {
                int m = mo + i + mrow;
                int n = no + tj * 16 + ncol;
                float s = sacc[tj][i] * scale + mp[i][kv0 + tj * 16];
                float p = expf(s);
                Ps[m * 64 + n] = f2bf(p);
                float sum = p;   // reduce across the 16 lanes of this half-wave
                sum += __shfl_xor(sum, 1);
                sum += __shfl_xor(sum, 2);
                sum += __shfl_xor(sum, 4);
                sum += __shfl_xor(sum, 8);
                if ((lane & 15) == 0) atomicAdd(&rowsum[m], sum);
            }
        }
        __syncthreads();

        // O += P V
        bf16x16 pa0 = load_a_frag(Ps, 64, mo, 0,  lane);
        bf16x16 pa1 = load_a_frag(Ps, 64, mo, 32, lane);
#pragma unroll
        for (int tj = 0; tj < 2; ++tj) {
            bf16x16 vb0 = load_b_frag(Vt, 64, no + tj * 16, 0,  lane);
            oacc[tj] = wmma_bf16(pa0, vb0, oacc[tj]);
            bf16x16 vb1 = load_b_frag(Vt, 64, no + tj * 16, 32, lane);
            oacc[tj] = wmma_bf16(pa1, vb1, oacc[tj]);
        }
    }
    __syncthreads();

    // epilogue: divide by (sum + 1e-10), write [B,S,H*Dh] so final GEMM consumes it directly
#pragma unroll
    for (int tj = 0; tj < 2; ++tj)
#pragma unroll
        for (int i = 0; i < 8; ++i) {
            int m = mo + i + mrow;
            int d = no + tj * 16 + ncol;
            float inv = 1.0f / (rowsum[m] + 1e-10f);
            Ob[(size_t)((size_t)blockIdx.y / H * S + q0 + m) * Dm + h * 64 + d] =
                f2bf(oacc[tj][i] * inv);
        }
}

// ---------- launch ----------

extern "C" void kernel_launch(void* const* d_in, const int* in_sizes, int n_in,
                              void* d_out, int out_size, void* d_ws, size_t ws_size,
                              hipStream_t stream) {
    const float* x    = (const float*)d_in[0];
    const float* y    = (const float*)d_in[1];
    const float* mask = (const float*)d_in[2];
    const float* Wq   = (const float*)d_in[3];
    const float* Wk   = (const float*)d_in[4];
    const float* Wv   = (const float*)d_in[5];
    const float* Wo   = (const float*)d_in[6];

    const int B = 2, S = 2048, Dm = 1024, H = 16;
    const size_t nx = (size_t)B * S * Dm;   // 4,194,304
    const size_t nw = (size_t)Dm * Dm;      // 1,048,576

    u16* xb  = (u16*)d_ws;
    u16* yb  = xb + nx;
    u16* wqb = yb + nx;
    u16* wkb = wqb + nw;
    u16* wvb = wkb + nw;
    u16* wob = wvb + nw;
    u16* qb  = wob + nw;
    u16* kb  = qb + nx;
    u16* vb  = kb + nx;
    u16* aob = vb + nx;   // total ~56 MB of d_ws

    const int blk = 256;
    cvt_bf16<<<(int)(nx / 4 + blk - 1) / blk, blk, 0, stream>>>(x,  xb,  (int)(nx / 4));
    cvt_bf16<<<(int)(nx / 4 + blk - 1) / blk, blk, 0, stream>>>(y,  yb,  (int)(nx / 4));
    cvt_bf16<<<(int)(nw / 4 + blk - 1) / blk, blk, 0, stream>>>(Wq, wqb, (int)(nw / 4));
    cvt_bf16<<<(int)(nw / 4 + blk - 1) / blk, blk, 0, stream>>>(Wk, wkb, (int)(nw / 4));
    cvt_bf16<<<(int)(nw / 4 + blk - 1) / blk, blk, 0, stream>>>(Wv, wvb, (int)(nw / 4));
    cvt_bf16<<<(int)(nw / 4 + blk - 1) / blk, blk, 0, stream>>>(Wo, wob, (int)(nw / 4));

    dim3 gg(Dm / 128, (B * S) / 128);   // (8, 32)
    gemm_bf16_nt<u16><<<gg, blk, 0, stream>>>(yb, wqb, qb, B * S, Dm, Dm);
    gemm_bf16_nt<u16><<<gg, blk, 0, stream>>>(xb, wkb, kb, B * S, Dm, Dm);
    gemm_bf16_nt<u16><<<gg, blk, 0, stream>>>(xb, wvb, vb, B * S, Dm, Dm);

    attn64<<<dim3(S / 64, B * H), blk, 0, stream>>>(qb, kb, vb, mask, aob);

    gemm_bf16_nt<float><<<gg, blk, 0, stream>>>(aob, wob, (float*)d_out, B * S, Dm, Dm);
}
